// GraphColorizer_11123965296666
// MI455X (gfx1250) — compile-verified, matrix operand order checked by reading, not run on hardware
//
#include <hip/hip_runtime.h>
#include <cstdint>
#include <cstddef>

// Problem constants (from reference)
#define NV   32768   // n_vertices
#define EMB  512     // embedding size
#define DEG  32      // padded degree == wave32 width
#define NC   50      // colors
#define H3R  281     // fc3 raw output
#define H3P  320     // fc3 padded to multiple of 64
#define CP   64      // colors padded to multiple of 64
#define NEG_INF (-1e30f)

typedef __attribute__((ext_vector_type(16))) _Float16 v16h;
typedef __attribute__((ext_vector_type(8)))  float    v8f;

static __device__ __forceinline__ float lrelu(float x) { return x > 0.f ? x : 0.01f * x; }

// ---------------- conversion / packing kernels ----------------
__global__ void f32_to_f16_kernel(const float* __restrict__ src, _Float16* __restrict__ dst, int n) {
  int i = blockIdx.x * blockDim.x + threadIdx.x;
  if (i < n) dst[i] = (_Float16)src[i];
}

// Bt[n][k] = (n<Nsrc && k<Ksrc) ? W[k][n] : 0   (dst is [Ndst][Kdst] f16, zero-padded)
__global__ void pack_bt_kernel(const float* __restrict__ src, _Float16* __restrict__ dst,
                               int Ksrc, int Nsrc, int Kdst, int Ndst) {
  int i = blockIdx.x * blockDim.x + threadIdx.x;
  if (i >= Ndst * Kdst) return;
  int n = i / Kdst, k = i % Kdst;
  dst[i] = (n < Nsrc && k < Ksrc) ? (_Float16)src[k * Nsrc + n] : (_Float16)0.f;
}

__global__ void pack_bias_kernel(const float* __restrict__ src, float* __restrict__ dst,
                                 int nsrc, int ndst) {
  int i = blockIdx.x * blockDim.x + threadIdx.x;
  if (i < ndst) dst[i] = (i < nsrc) ? src[i] : 0.f;
}

// ---------------- WMMA f16 GEMM: C = act(A[M,K] @ Bt[N,K]^T + bias) ----------------
// Block = 256 threads = 8 waves; 128x64 output tile; K stepped by 32.
// Register-staged LDS double buffering: global loads for tile k+1 issue before the
// WMMAs of tile k, so load latency hides behind matrix work; one barrier per step.
// 80B LDS row stride keeps the per-lane ds_load_b128 fragment reads bank-conflict-free.
__global__ __launch_bounds__(256)
void gemm_f16_kernel(const _Float16* __restrict__ A, const _Float16* __restrict__ Bt,
                     const float* __restrict__ bias, void* __restrict__ Cout,
                     int M, int K, int Nout, int do_relu, int out_f16) {
  __shared__ uint4 sA4[2][128 * 5];   // 2 x (128 rows x 4 chunks + pad)
  __shared__ uint4 sB4[2][64 * 5];    // 2 x ( 64 rows x 4 chunks + pad)
  const uint4* Aq = (const uint4*)A;  // K/8 16B chunks per row
  const uint4* Bq = (const uint4*)Bt;

  int m0 = blockIdx.y * 128;
  int n0 = blockIdx.x * 64;
  int tid = threadIdx.x, wave = tid >> 5, lane = tid & 31;
  int half = lane >> 4, l16 = lane & 15;
  int Kc = K >> 3;
  int rr = tid >> 2, cc = tid & 3;    // fill coordinates: rows rr / rr+64 (A), rr (B)

  v8f acc[4];
#pragma unroll
  for (int t = 0; t < 4; t++)
#pragma unroll
    for (int e = 0; e < 8; e++) acc[t][e] = 0.f;

  union HV { uint4 q[2]; v16h h; };

  // Prologue: stage tile 0
  {
    uint4 a0 = Aq[(size_t)(m0 + rr) * Kc + cc];
    uint4 a1 = Aq[(size_t)(m0 + rr + 64) * Kc + cc];
    uint4 b0 = Bq[(size_t)(n0 + rr) * Kc + cc];
    sA4[0][rr * 5 + cc] = a0;
    sA4[0][(rr + 64) * 5 + cc] = a1;
    sB4[0][rr * 5 + cc] = b0;
  }
  __syncthreads();

  int p = 0;
  int arow = wave * 16 + l16;
  for (int k0 = 0; k0 < K; k0 += 32) {
    bool more = (k0 + 32 < K);
    int kcn = (k0 + 32) >> 3;
    uint4 na0 = {}, na1 = {}, nb0 = {};
    if (more) {                          // issue next tile's global loads early
      na0 = Aq[(size_t)(m0 + rr) * Kc + kcn + cc];
      na1 = Aq[(size_t)(m0 + rr + 64) * Kc + kcn + cc];
      nb0 = Bq[(size_t)(n0 + rr) * Kc + kcn + cc];
      if (k0 + 64 < K)                   // CDNA5 global_prefetch_b8 two tiles ahead
        __builtin_prefetch(&Aq[(size_t)(m0 + rr) * Kc + kcn + 4 + cc], 0, 1);
    }

    const uint4* bufA = sA4[p];
    const uint4* bufB = sB4[p];
    HV av;                               // A fragment: 16B chunks half and 2+half (table k order)
    av.q[0] = bufA[arow * 5 + half];
    av.q[1] = bufA[arow * 5 + 2 + half];
#pragma unroll
    for (int nt = 0; nt < 4; nt++) {
      HV bv;
      int brow = nt * 16 + l16;
      bv.q[0] = bufB[brow * 5 + half];
      bv.q[1] = bufB[brow * 5 + 2 + half];
      acc[nt] = __builtin_amdgcn_wmma_f32_16x16x32_f16(
          false, av.h, false, bv.h, (short)0, acc[nt], false, false);
    }

    if (more) {                          // commit staged regs to the other buffer
      sA4[p ^ 1][rr * 5 + cc] = na0;
      sA4[p ^ 1][(rr + 64) * 5 + cc] = na1;
      sB4[p ^ 1][rr * 5 + cc] = nb0;
    }
    __syncthreads();
    p ^= 1;
  }

  // Epilogue: bias + leaky_relu; D layout: VGPR i -> row i + 8*half, lane&15 -> col
#pragma unroll
  for (int nt = 0; nt < 4; nt++) {
    int gn = n0 + nt * 16 + l16;
    float bb = bias[gn];
#pragma unroll
    for (int i = 0; i < 8; i++) {
      int gm = m0 + wave * 16 + half * 8 + i;
      float v = acc[nt][i] + bb;
      if (do_relu) v = lrelu(v);
      if (out_f16) ((_Float16*)Cout)[(size_t)gm * Nout + gn] = (_Float16)v;
      else         ((float*)Cout)[(size_t)gm * Nout + gn] = v;
    }
  }
}

// ---------------- sequential greedy-coloring scan: 1 wave32, colors in 128KB LDS ----------------
static __device__ __forceinline__ float gumbelf(int i, int c) {
  unsigned x = (unsigned)(i * 64 + c) * 0x9E3779B1u + 0x85ebca6bu;
  x ^= x >> 16; x *= 0x7feb352du; x ^= x >> 15; x *= 0x846ca68bu; x ^= x >> 16;
  float u = (x >> 8) * (1.0f / 16777216.0f);
  u = u * 0.999998f + 1e-6f;
  return -__logf(-__logf(u));
}

// Single wave: LDS ops complete in program order per wave, so no workgroup barriers
// are required -- only compiler scheduling fences (zero hardware cost).
__global__ void scan_kernel(const int* __restrict__ order, const int* __restrict__ adj,
                            const float* __restrict__ logits, int* __restrict__ colors_g,
                            float* __restrict__ out_colors, float* __restrict__ conf_out) {
  extern __shared__ int colors_s[];   // NV ints = 128KB (fits CDNA5 320KB WGP LDS)
  __shared__ unsigned sm[2];
  int lane = threadIdx.x;

  for (int u = lane; u < NV; u += 32) colors_s[u] = -1;
  __builtin_amdgcn_wave_barrier();
  if (lane == 0) colors_s[order[0]] = 0;
  __builtin_amdgcn_wave_barrier();

  float logp = 0.f;
  const float LOG_EPS = __logf(1e-8f);
  int v  = order[1];
  int an = adj[(size_t)v * DEG + lane];          // lane <-> neighbor slot (DEG==32)

  for (int i = 1; i < NV; i++) {
    int vn = (i + 1 < NV) ? order[i + 1] : 0;    // software-pipelined next-vertex fetch
    int an_next = adj[(size_t)vn * DEG + lane];  // (adj is 4MB -> L2 resident)

    if (lane < 2) sm[lane] = 0;
    __builtin_amdgcn_wave_barrier();
    int nc = colors_s[an];
    if (nc >= 0) atomicOr(&sm[nc >> 5], 1u << (nc & 31));
    __builtin_amdgcn_wave_barrier();
    unsigned mk0 = sm[0], mk1 = sm[1];

    int c0 = lane, c1 = lane + 32;
    float l0 = ((mk0 >> c0) & 1u) ? NEG_INF : logits[(size_t)v * CP + c0];
    float l1 = NEG_INF;
    if (c1 < NC) l1 = ((mk1 >> (c1 & 31)) & 1u) ? NEG_INF : logits[(size_t)v * CP + c1];

    float k0 = l0 + gumbelf(i, c0);
    float k1 = (c1 < NC) ? l1 + gumbelf(i, c1) : NEG_INF;
    float key; int idx;
    if (k0 >= k1) { key = k0; idx = c0; } else { key = k1; idx = c1; }
    float mx = fmaxf(l0, l1);
#pragma unroll
    for (int off = 16; off > 0; off >>= 1) {     // butterfly argmax + max
      float ok = __shfl_xor(key, off, 32);
      int   oi = __shfl_xor(idx, off, 32);
      if (ok > key) { key = ok; idx = oi; }
      mx = fmaxf(mx, __shfl_xor(mx, off, 32));
    }
    // Critical inter-iteration value: commit the chosen color immediately.
    if (lane == 0) colors_s[v] = idx;
    __builtin_amdgcn_wave_barrier();

    // Off-critical-path bookkeeping: softmax log-prob of the chosen color.
    float e = 0.f;
    if (l0 > 0.5f * NEG_INF) e += __expf(l0 - mx);
    if (l1 > 0.5f * NEG_INF) e += __expf(l1 - mx);
#pragma unroll
    for (int off = 16; off > 0; off >>= 1) e += __shfl_xor(e, off, 32);
    float lsrc = (idx >= 32) ? l1 : l0;          // idx is wave-uniform here
    float l_ch = __shfl(lsrc, idx & 31, 32);
    if (lane == 0) {
      float pch = __expf(l_ch - mx) / e;
      logp += __logf(pch + 1e-8f) - LOG_EPS;
    }
    v = vn; an = an_next;
  }
  __builtin_amdgcn_wave_barrier();
  for (int u = lane; u < NV; u += 32) {
    int c = colors_s[u];
    colors_g[u] = c;
    out_colors[u] = (float)c;
  }
  if (lane == 0) *conf_out = logp / (float)NV;   // confidence = logp / N
}

// ---------------- stats ----------------
__global__ void init_stats_kernel(unsigned* st) {
  if (blockIdx.x == 0 && threadIdx.x < 3) st[threadIdx.x] = 0;  // viol, mask0, mask1
}

__global__ void stats_kernel(const int* __restrict__ colors_g, const int* __restrict__ adj,
                             unsigned* __restrict__ st) {
  int v = blockIdx.x * blockDim.x + threadIdx.x;
  if (v >= NV) return;
  int cv = colors_g[v];
  if (cv >= 0 && cv < 64) atomicOr(&st[1 + (cv >> 5)], 1u << (cv & 31));
  int cnt = 0;
#pragma unroll 4
  for (int d = 0; d < DEG; d++) cnt += (colors_g[adj[(size_t)v * DEG + d]] == cv) ? 1 : 0;
  if (cnt) atomicAdd(&st[0], (unsigned)cnt);
}

__global__ void final_kernel(const unsigned* __restrict__ st, const float* __restrict__ conf,
                             float* __restrict__ out) {
  int n_used = __popc(st[1]) + __popc(st[2]);
  float ratio = (float)st[0] / (float)((size_t)NV * DEG);
  float cost = (float)n_used + 100.f * ratio;
  out[NV] = cost * (*conf);
}

// ---------------- host launcher ----------------
extern "C" void kernel_launch(void* const* d_in, const int* in_sizes, int n_in,
                              void* d_out, int out_size, void* d_ws, size_t ws_size,
                              hipStream_t stream) {
  const float* emb   = (const float*)d_in[0];
  const int*   adj   = (const int*)d_in[1];
  const int*   order = (const int*)d_in[2];
  const float* W1 = (const float*)d_in[3]; const float* b1 = (const float*)d_in[4];
  const float* W2 = (const float*)d_in[5]; const float* b2 = (const float*)d_in[6];
  const float* W3 = (const float*)d_in[7]; const float* b3 = (const float*)d_in[8];
  const float* W4 = (const float*)d_in[9]; const float* b4 = (const float*)d_in[10];
  (void)in_sizes; (void)n_in; (void)out_size; (void)ws_size;

  char* ws = (char*)d_ws;
  size_t off = 0;
  auto alloc = [&](size_t bytes) -> void* {
    void* p = ws + off;
    off = (off + bytes + 255) & ~(size_t)255;
    return p;
  };
  _Float16* buf0 = (_Float16*)alloc((size_t)NV * EMB * 2);  // Xh, then H3h (reuse)
  _Float16* buf1 = (_Float16*)alloc((size_t)NV * EMB * 2);  // H1h, then logits f32 (reuse)
  _Float16* buf2 = (_Float16*)alloc((size_t)NV * EMB * 2);  // H2h
  _Float16* W1t  = (_Float16*)alloc((size_t)EMB * EMB * 2); // [N][K] transposed weights
  _Float16* W2t  = (_Float16*)alloc((size_t)EMB * EMB * 2);
  _Float16* W3t  = (_Float16*)alloc((size_t)H3P * EMB * 2);
  _Float16* W4t  = (_Float16*)alloc((size_t)CP * H3P * 2);
  float*    b3p  = (float*)alloc(H3P * 4);
  float*    b4p  = (float*)alloc(CP * 4);
  int*      colors_g = (int*)alloc((size_t)NV * 4);
  float*    conf_p   = (float*)alloc(4);
  unsigned* st       = (unsigned*)alloc(16);

  const int T = 256;
  f32_to_f16_kernel<<<(NV * EMB + T - 1) / T, T, 0, stream>>>(emb, buf0, NV * EMB);
  // transposed, zero-padded weights: Bt[n][k]
  pack_bt_kernel<<<(EMB * EMB + T - 1) / T, T, 0, stream>>>(W1, W1t, EMB, EMB, EMB, EMB);
  pack_bt_kernel<<<(EMB * EMB + T - 1) / T, T, 0, stream>>>(W2, W2t, EMB, EMB, EMB, EMB);
  pack_bt_kernel<<<(H3P * EMB + T - 1) / T, T, 0, stream>>>(W3, W3t, EMB, H3R, EMB, H3P);
  pack_bt_kernel<<<(CP * H3P + T - 1) / T, T, 0, stream>>>(W4, W4t, H3R, NC, H3P, CP);
  pack_bias_kernel<<<(H3P + T - 1) / T, T, 0, stream>>>(b3, b3p, H3R, H3P);
  pack_bias_kernel<<<(CP + T - 1) / T, T, 0, stream>>>(b4, b4p, NC, CP);

  // Batched MLP: all sequential-scan-independent compute as WMMA GEMMs.
  gemm_f16_kernel<<<dim3(EMB / 64, NV / 128), 256, 0, stream>>>(
      buf0, W1t, b1, buf1, NV, EMB, EMB, 1, 1);                       // H1
  gemm_f16_kernel<<<dim3(EMB / 64, NV / 128), 256, 0, stream>>>(
      buf1, W2t, b2, buf2, NV, EMB, EMB, 1, 1);                       // H2
  gemm_f16_kernel<<<dim3(H3P / 64, NV / 128), 256, 0, stream>>>(
      buf2, W3t, b3p, buf0, NV, EMB, H3P, 1, 1);                      // H3 (padded)
  float* logits = (float*)buf1;
  gemm_f16_kernel<<<dim3(CP / 64, NV / 128), 256, 0, stream>>>(
      buf0, W4t, b4p, logits, NV, H3P, CP, 0, 0);                     // logits f32

  // Sequential color assignment: single wave, colors live in 128KB of WGP LDS.
  scan_kernel<<<1, 32, (size_t)NV * sizeof(int), stream>>>(
      order, adj, logits, colors_g, (float*)d_out, conf_p);

  init_stats_kernel<<<1, 32, 0, stream>>>(st);
  stats_kernel<<<NV / 256, 256, 0, stream>>>(colors_g, adj, st);
  final_kernel<<<1, 1, 0, stream>>>(st, conf_p, (float*)d_out);
}
